// Transformer_37692632990168
// MI455X (gfx1250) — matrix operand order repaired and checked by reading, not compiled, not run on hardware
//
#include <hip/hip_runtime.h>
#include <math.h>

// ---------------------------------------------------------------------------
// Transformer forward for MI455X (gfx1250, wave32, WMMA bf16 -> f32 accum)
// ---------------------------------------------------------------------------

#define B_  2
#define S_  1024
#define E_  1024
#define H_  16
#define DK_ 64
#define F_  4096
#define L_  4
#define V_  32000
#define BS_ (B_ * S_)

typedef __attribute__((ext_vector_type(16))) __bf16 v16bf;
typedef __attribute__((ext_vector_type(8)))  float  v8f;

union ABpack { v16bf v; unsigned int u[8]; unsigned short h[16]; };
union Q4pack { uint4 q; unsigned short h[8]; unsigned int u[4]; };

static __device__ inline v8f wmma_bf16(const v16bf& a, const v16bf& b, const v8f& c) {
    return __builtin_amdgcn_wmma_f32_16x16x32_bf16(false, a, false, b, (short)0, c,
                                                   false, false);
}

static __device__ inline unsigned short f2bf(float f) {
    unsigned int u = __float_as_uint(f);
    unsigned int r = 0x7FFFu + ((u >> 16) & 1u);   // round-to-nearest-even
    return (unsigned short)((u + r) >> 16);
}

// A-operand K index pattern for 16x16x32 bf16 WMMA (per lane half)
static __device__ inline int a_kpat(int v, int half) {
    return 16 * half + 2 * v;
}

// ---------------------------------------------------------------------------
// Weight conversion kernels
// ---------------------------------------------------------------------------
__global__ void to_bf16_kernel(const float* __restrict__ src,
                               unsigned short* __restrict__ dst, size_t n) {
    size_t i = (size_t)blockIdx.x * blockDim.x + threadIdx.x;
    if (i < n) dst[i] = f2bf(src[i]);
}

// Wq/Wk/Wv [L,H,E,DK] -> [L, E, H*DK]
__global__ void conv_qkv_kernel(const float* __restrict__ W,
                                unsigned short* __restrict__ dst) {
    size_t idx = (size_t)blockIdx.x * blockDim.x + threadIdx.x;
    if (idx >= (size_t)L_ * E_ * (H_ * DK_)) return;
    int c = (int)(idx & 1023);          // h*64 + k
    int e = (int)((idx >> 10) & 1023);
    int l = (int)(idx >> 20);
    int h = c >> 6, k = c & 63;
    size_t src = (((size_t)(l * H_ + h) * E_) + e) * DK_ + k;
    dst[idx] = f2bf(W[src]);
}

// ---------------------------------------------------------------------------
// Embedding gather: emb f32 + bf16
// ---------------------------------------------------------------------------
__global__ void embed_kernel(const int* __restrict__ tokens,
                             const float* __restrict__ table,
                             float* __restrict__ emb,
                             unsigned short* __restrict__ emb_bf) {
    size_t idx = (size_t)blockIdx.x * blockDim.x + threadIdx.x;
    if (idx >= (size_t)BS_ * E_) return;
    int row = (int)(idx >> 10);
    int e   = (int)(idx & 1023);
    int tok = tokens[row];
    float v = table[(size_t)tok * E_ + e];
    emb[idx] = v;
    emb_bf[idx] = f2bf(v);
}

// ---------------------------------------------------------------------------
// Generic bf16 GEMM with WMMA: C[M,N] = A[M,K] @ B[K,N] (+ epilogue)
// mode 0: C = A@B                          (f32 out)
// mode 1: C = R + A@B, also Cbf = bf16(C)  (residual fuse)
// mode 2: C = A@B + bias[col]              (R = bias vector)
// Block: 256 threads (8 waves), tile 128(M) x 128(N), K-step 32,
// double-buffered LDS. Each wave computes 32x64 (8 WMMAs / K-step, A reused).
// ---------------------------------------------------------------------------
__global__ __launch_bounds__(256)
void gemm_bf16_kernel(const unsigned short* __restrict__ A,
                      const unsigned short* __restrict__ Bm,
                      float* __restrict__ C,
                      const float* __restrict__ R,
                      unsigned short* __restrict__ Cbf,
                      int M, int N, int K, int mode) {
    __shared__ unsigned short As[2][128 * 36];   // [m][k], stride 36
    __shared__ unsigned short Bs[2][128 * 36];   // [n][k] transposed, K-pairs packed

    const int tid  = threadIdx.x;
    const int lane = tid & 31;
    const int wave = tid >> 5;                   // 0..7
    const int wm   = wave & 3;                   // 4 row groups of 32
    const int wn   = wave >> 2;                  // 2 col groups of 64
    const int half = lane >> 4;
    const int ln16 = lane & 15;

    const int rowBase = blockIdx.y * 128;
    const int colBase = blockIdx.x * 128;

    v8f acc[2][4];
#pragma unroll
    for (int i = 0; i < 2; ++i)
#pragma unroll
        for (int j = 0; j < 4; ++j)
            acc[i][j] = (v8f){0.f, 0.f, 0.f, 0.f, 0.f, 0.f, 0.f, 0.f};

    // staging assignments (256 threads)
    const int arow  = tid >> 1;                  // 0..127
    const int achlf = tid & 1;                   // 16-col half of the 32 K cols
    const int brow2 = (tid & 15) * 2;            // even K row 0..30
    const int bchnk = tid >> 4;                  // 0..15 -> 8-col chunk

    const int nk = K >> 5;

    // ---- prologue: fetch + store tile 0 ----
    uint4 ga0, ga1, gb0, gb1;
    {
        const uint4* g = (const uint4*)(A + (size_t)(rowBase + arow) * K + 16 * achlf);
        ga0 = g[0]; ga1 = g[1];
        const unsigned short* bp0 = Bm + (size_t)brow2 * N + colBase + 8 * bchnk;
        gb0 = *(const uint4*)bp0;
        gb1 = *(const uint4*)(bp0 + N);
    }
    {
        uint2* s = (uint2*)&As[0][arow * 36 + 16 * achlf];
        s[0] = make_uint2(ga0.x, ga0.y);
        s[1] = make_uint2(ga0.z, ga0.w);
        s[2] = make_uint2(ga1.x, ga1.y);
        s[3] = make_uint2(ga1.z, ga1.w);
        Q4pack x0, x1; x0.q = gb0; x1.q = gb1;
#pragma unroll
        for (int e = 0; e < 8; ++e) {
            unsigned int pk = (unsigned int)x0.h[e] | ((unsigned int)x1.h[e] << 16);
            *(unsigned int*)&Bs[0][(8 * bchnk + e) * 36 + brow2] = pk;
        }
    }

    int cur = 0;
    for (int kt = 0; kt < nk; ++kt) {
        __syncthreads();                         // buf[cur] ready for all waves

        const bool has_next = (kt + 1) < nk;
        if (has_next) {                          // fetch next tile into registers
            int k0 = (kt + 1) << 5;
            const uint4* g = (const uint4*)(A + (size_t)(rowBase + arow) * K + k0 + 16 * achlf);
            ga0 = g[0]; ga1 = g[1];
            const unsigned short* bp0 = Bm + (size_t)(k0 + brow2) * N + colBase + 8 * bchnk;
            gb0 = *(const uint4*)bp0;
            gb1 = *(const uint4*)(bp0 + N);
            if (kt + 2 < nk)
                __builtin_prefetch(A + (size_t)(rowBase + arow) * K + ((kt + 2) << 5), 0, 3);
        }

        // ---- WMMA compute from buf[cur]: 2 A-tiles x 4 B-tiles ----
        ABpack a[2], b[4];
#pragma unroll
        for (int i = 0; i < 2; ++i) {
            int m = wm * 32 + 16 * i + ln16;
#pragma unroll
            for (int v = 0; v < 8; ++v)
                a[i].u[v] = *(const unsigned int*)&As[cur][m * 36 + a_kpat(v, half)];
        }
#pragma unroll
        for (int j = 0; j < 4; ++j) {
            int n = wn * 64 + 16 * j + ln16;
#pragma unroll
            for (int v = 0; v < 8; ++v)
                b[j].u[v] = *(const unsigned int*)&Bs[cur][n * 36 + 16 * half + 2 * v];
        }
#pragma unroll
        for (int i = 0; i < 2; ++i)
#pragma unroll
            for (int j = 0; j < 4; ++j)
                acc[i][j] = wmma_bf16(a[i].v, b[j].v, acc[i][j]);

        if (has_next) {                          // store next tile into other buffer
            int nxt = cur ^ 1;
            uint2* s = (uint2*)&As[nxt][arow * 36 + 16 * achlf];
            s[0] = make_uint2(ga0.x, ga0.y);
            s[1] = make_uint2(ga0.z, ga0.w);
            s[2] = make_uint2(ga1.x, ga1.y);
            s[3] = make_uint2(ga1.z, ga1.w);
            Q4pack x0, x1; x0.q = gb0; x1.q = gb1;
#pragma unroll
            for (int e = 0; e < 8; ++e) {
                unsigned int pk = (unsigned int)x0.h[e] | ((unsigned int)x1.h[e] << 16);
                *(unsigned int*)&Bs[nxt][(8 * bchnk + e) * 36 + brow2] = pk;
            }
        }
        cur ^= 1;
    }

    // ---- epilogue ----
#pragma unroll
    for (int i = 0; i < 2; ++i)
#pragma unroll
        for (int j = 0; j < 4; ++j)
#pragma unroll
            for (int r = 0; r < 8; ++r) {
                int gr = rowBase + wm * 32 + 16 * i + (half << 3) + r;
                int gc = colBase + wn * 64 + 16 * j + ln16;
                float val = acc[i][j][r];
                size_t o = (size_t)gr * N + gc;
                if (mode == 1)      val += R[o];
                else if (mode == 2) val += R[gc];
                C[o] = val;
                if (mode == 1) Cbf[o] = f2bf(val);
            }
}

// ---------------------------------------------------------------------------
// RoPE on Q,K (gemm layout [BS, H*DK]) -> bf16 [B,H,S,DK]; V -> bf16 [B,H,DK,S]
// ---------------------------------------------------------------------------
__global__ void rope_vt_kernel(const float* __restrict__ Qf,
                               const float* __restrict__ Kf,
                               const float* __restrict__ Vf,
                               unsigned short* __restrict__ Qbf,
                               unsigned short* __restrict__ Kbf,
                               unsigned short* __restrict__ Vt) {
    size_t idx = (size_t)blockIdx.x * blockDim.x + threadIdx.x;
    if (idx >= (size_t)B_ * S_ * H_ * DK_) return;
    int k = (int)(idx & 63);
    int h = (int)((idx >> 6) & 15);
    int s = (int)((idx >> 10) & 1023);
    int b = (int)(idx >> 20);

    size_t row = (size_t)b * S_ + s;
    const float* qr = Qf + row * (H_ * DK_) + h * DK_;
    const float* kr = Kf + row * (H_ * DK_) + h * DK_;

    float theta = __powf(10000.f, (-2.f / 64.f) * (float)(k & 31));
    float ang = (float)s * theta;
    float c = __cosf(ang);
    float sn = __sinf(ang) * ((k < 32) ? -1.f : 1.f);

    float qv = qr[k], kv = kr[k];
    float qs = (k < 32) ? qr[k + 32] : qr[k - 32];
    float ks = (k < 32) ? kr[k + 32] : kr[k - 32];

    size_t o = (((size_t)(b * H_ + h) * S_) + s) * DK_ + k;
    Qbf[o] = f2bf(c * qv + sn * qs);
    Kbf[o] = f2bf(c * kv + sn * ks);
    Vt[((size_t)(b * H_ + h) * DK_ + k) * S_ + s] = f2bf(Vf[row * (H_ * DK_) + h * DK_ + k]);
}

// ---------------------------------------------------------------------------
// Flash attention: one wave per (b, h, 16-query tile). Online softmax over
// 32-key blocks; scores and P@V both via WMMA bf16.
// ---------------------------------------------------------------------------
__global__ __launch_bounds__(32)
void attention_kernel(const unsigned short* __restrict__ Qbf,
                      const unsigned short* __restrict__ Kbf,
                      const unsigned short* __restrict__ Vt,
                      unsigned short* __restrict__ Obf) {
    __shared__ unsigned short Ps[16 * 36];    // P tile bf16 [16 rows][32 cols]

    const int qt = blockIdx.x;
    const int h  = blockIdx.y;
    const int b  = blockIdx.z;
    const int lane = threadIdx.x;
    const int ln16 = lane & 15;
    const int half = lane >> 4;
    const float inv_sqrt = 0.125f;  // 1/sqrt(64)

    const size_t base_qk = (size_t)(b * H_ + h) * S_ * DK_;
    const size_t base_v  = (size_t)(b * H_ + h) * DK_ * S_;

    // Q operands (16x64 -> two 16x32 A tiles)
    ABpack q0, q1;
    {
        const unsigned short* Qrow = Qbf + base_qk + (size_t)(qt * 16 + ln16) * DK_;
#pragma unroll
        for (int v = 0; v < 8; ++v) {
            int k = a_kpat(v, half);
            q0.u[v] = *(const unsigned int*)&Qrow[k];
            q1.u[v] = *(const unsigned int*)&Qrow[32 + k];
        }
    }

    float m_run[8], l_run[8];
    v8f o[4];
#pragma unroll
    for (int r = 0; r < 8; ++r) { m_run[r] = -INFINITY; l_run[r] = 0.f; }
#pragma unroll
    for (int j = 0; j < 4; ++j) o[j] = (v8f){0.f, 0.f, 0.f, 0.f, 0.f, 0.f, 0.f, 0.f};

    const int ktmax = (qt * 16 + 15) >> 5;
    for (int kt = 0; kt <= ktmax; ++kt) {
        // ---- scores: two 16x16 tiles (keys kt*32 .. kt*32+31) ----
        float sc[2][8];
#pragma unroll
        for (int t = 0; t < 2; ++t) {
            int key = kt * 32 + t * 16 + ln16;
            const unsigned short* Krow = Kbf + base_qk + (size_t)key * DK_;
            ABpack kb0, kb1;
#pragma unroll
            for (int v = 0; v < 8; ++v) {
                int kk = 16 * half + 2 * v;
                kb0.u[v] = *(const unsigned int*)&Krow[kk];
                kb1.u[v] = *(const unsigned int*)&Krow[32 + kk];
            }
            v8f s = (v8f){0.f, 0.f, 0.f, 0.f, 0.f, 0.f, 0.f, 0.f};
            s = wmma_bf16(q0.v, kb0.v, s);
            s = wmma_bf16(q1.v, kb1.v, s);
#pragma unroll
            for (int r = 0; r < 8; ++r) {
                int row = qt * 16 + (half << 3) + r;
                float val = s[r] * inv_sqrt;
                sc[t][r] = (key > row) ? -INFINITY : val;
            }
        }

        // ---- online softmax update ----
        float alpha[8], mnew[8];
#pragma unroll
        for (int r = 0; r < 8; ++r) {
            float x = fmaxf(sc[0][r], sc[1][r]);
#pragma unroll
            for (int off = 1; off < 16; off <<= 1)
                x = fmaxf(x, __shfl_xor(x, off));
            mnew[r]  = fmaxf(m_run[r], x);
            alpha[r] = __expf(m_run[r] - mnew[r]);
            m_run[r] = mnew[r];
        }
        float p0[8], p1[8];
#pragma unroll
        for (int r = 0; r < 8; ++r) {
            p0[r] = __expf(sc[0][r] - mnew[r]);
            p1[r] = __expf(sc[1][r] - mnew[r]);
            float rs = p0[r] + p1[r];
#pragma unroll
            for (int off = 1; off < 16; off <<= 1)
                rs += __shfl_xor(rs, off);
            l_run[r] = l_run[r] * alpha[r] + rs;
        }
#pragma unroll
        for (int j = 0; j < 4; ++j)
#pragma unroll
            for (int r = 0; r < 8; ++r)
                o[j][r] *= alpha[r];

        // ---- re-lay P (C layout) to A layout through LDS ----
        __syncthreads();
#pragma unroll
        for (int r = 0; r < 8; ++r) {
            int rl = (half << 3) + r;
            Ps[rl * 36 + ln16]      = f2bf(p0[r]);
            Ps[rl * 36 + 16 + ln16] = f2bf(p1[r]);
        }
        __syncthreads();
        ABpack pa;
#pragma unroll
        for (int v = 0; v < 8; ++v)
            pa.u[v] = *(const unsigned int*)&Ps[ln16 * 36 + a_kpat(v, half)];

        // ---- O += P @ V  (V pre-transposed [DK][S]) ----
#pragma unroll
        for (int j = 0; j < 4; ++j) {
            int d = j * 16 + ln16;
            const unsigned short* Vrow = Vt + base_v + (size_t)d * S_ + kt * 32;
            ABpack vb;
#pragma unroll
            for (int v = 0; v < 8; ++v)
                vb.u[v] = *(const unsigned int*)&Vrow[16 * half + 2 * v];
            o[j] = wmma_bf16(pa.v, vb.v, o[j]);
        }
    }

    // ---- normalize + write bf16 [BS, H*DH] for the proj GEMM ----
#pragma unroll
    for (int j = 0; j < 4; ++j)
#pragma unroll
        for (int r = 0; r < 8; ++r) {
            int row = qt * 16 + (half << 3) + r;
            int col = h * DK_ + j * 16 + ln16;
            float val = o[j][r] / l_run[r];
            Obf[((size_t)b * S_ + row) * (H_ * DK_) + col] = f2bf(val);
        }
}

// ---------------------------------------------------------------------------
// RMSNorm (block per row) -> bf16
// ---------------------------------------------------------------------------
__global__ __launch_bounds__(256)
void rmsnorm_kernel(const float* __restrict__ X, const float* __restrict__ w,
                    unsigned short* __restrict__ Hbf) {
    __shared__ float red[256];
    int row = blockIdx.x;
    int tid = threadIdx.x;
    const float* xr = X + (size_t)row * E_;
    float s = 0.f;
    for (int i = tid; i < E_; i += 256) { float v = xr[i]; s += v * v; }
    red[tid] = s;
    __syncthreads();
    for (int off = 128; off > 0; off >>= 1) {
        if (tid < off) red[tid] += red[tid + off];
        __syncthreads();
    }
    float inv = rsqrtf(red[0] * (1.f / E_) + 1.1920929e-07f);
    for (int i = tid; i < E_; i += 256)
        Hbf[(size_t)row * E_ + i] = f2bf(xr[i] * inv * w[i]);
}

// ---------------------------------------------------------------------------
// prod = bf16( U * elu(G) )
// ---------------------------------------------------------------------------
__global__ void gate_prod_kernel(const float* __restrict__ U,
                                 const float* __restrict__ G,
                                 unsigned short* __restrict__ P, size_t n) {
    size_t i = (size_t)blockIdx.x * blockDim.x + threadIdx.x;
    if (i >= n) return;
    float g = G[i];
    float e = (g > 0.f) ? g : expm1f(g);
    P[i] = f2bf(U[i] * e);
}

// ---------------------------------------------------------------------------
// Host orchestration
// ---------------------------------------------------------------------------
static inline size_t align256(size_t x) { return (x + 255) & ~(size_t)255; }

extern "C" void kernel_launch(void* const* d_in, const int* in_sizes, int n_in,
                              void* d_out, int out_size, void* d_ws, size_t ws_size,
                              hipStream_t stream) {
    (void)in_sizes; (void)n_in; (void)out_size; (void)ws_size;
    const int*   tokens = (const int*)d_in[0];
    const float* table  = (const float*)d_in[1];
    const float* Wq     = (const float*)d_in[2];
    const float* Wk     = (const float*)d_in[3];
    const float* Wv     = (const float*)d_in[4];
    const float* Wproj  = (const float*)d_in[5];
    const float* nw     = (const float*)d_in[6];
    const float* Wup    = (const float*)d_in[7];
    const float* Wgate  = (const float*)d_in[8];
    const float* Wdown  = (const float*)d_in[9];
    const float* predW  = (const float*)d_in[10];
    const float* predb  = (const float*)d_in[11];
    float* out = (float*)d_out;

    char* ws = (char*)d_ws;
    size_t off = 0;
    auto alloc = [&](size_t bytes) { void* p = ws + off; off = align256(off + bytes); return p; };

    unsigned short* wq_bf = (unsigned short*)alloc((size_t)L_ * E_ * 1024 * 2);
    unsigned short* wk_bf = (unsigned short*)alloc((size_t)L_ * E_ * 1024 * 2);
    unsigned short* wv_bf = (unsigned short*)alloc((size_t)L_ * E_ * 1024 * 2);
    unsigned short* wp_bf = (unsigned short*)alloc((size_t)L_ * E_ * E_ * 2);
    unsigned short* wu_bf = (unsigned short*)alloc((size_t)L_ * E_ * F_ * 2);
    unsigned short* wg_bf = (unsigned short*)alloc((size_t)L_ * E_ * F_ * 2);
    unsigned short* wd_bf = (unsigned short*)alloc((size_t)L_ * F_ * E_ * 2);
    unsigned short* pw_bf = (unsigned short*)alloc((size_t)E_ * V_ * 2);
    float*          emb   = (float*)alloc((size_t)BS_ * E_ * 4);
    unsigned short* embbf = (unsigned short*)alloc((size_t)BS_ * E_ * 2);
    float*          Qf    = (float*)alloc((size_t)BS_ * 1024 * 4);
    float*          Kf    = (float*)alloc((size_t)BS_ * 1024 * 4);
    float*          Vf    = (float*)alloc((size_t)BS_ * 1024 * 4);
    unsigned short* Qbf   = (unsigned short*)alloc((size_t)BS_ * 1024 * 2);
    unsigned short* Kbf   = (unsigned short*)alloc((size_t)BS_ * 1024 * 2);
    unsigned short* Vtb   = (unsigned short*)alloc((size_t)BS_ * 1024 * 2);
    unsigned short* Obf   = (unsigned short*)alloc((size_t)BS_ * 1024 * 2);
    unsigned short* hbf   = (unsigned short*)alloc((size_t)BS_ * E_ * 2);
    float*          Uf    = (float*)alloc((size_t)BS_ * F_ * 4);
    float*          Gf    = (float*)alloc((size_t)BS_ * F_ * 4);
    unsigned short* pbf   = (unsigned short*)alloc((size_t)BS_ * F_ * 2);

    auto gblk = [](size_t n) { return (unsigned)((n + 255) / 256); };

    // ---- weight conversion (once) ----
    {
        size_t nq = (size_t)L_ * E_ * 1024;
        conv_qkv_kernel<<<gblk(nq), 256, 0, stream>>>(Wq, wq_bf);
        conv_qkv_kernel<<<gblk(nq), 256, 0, stream>>>(Wk, wk_bf);
        conv_qkv_kernel<<<gblk(nq), 256, 0, stream>>>(Wv, wv_bf);
        size_t np = (size_t)L_ * E_ * E_;
        to_bf16_kernel<<<gblk(np), 256, 0, stream>>>(Wproj, wp_bf, np);
        size_t nm = (size_t)L_ * E_ * F_;
        to_bf16_kernel<<<gblk(nm), 256, 0, stream>>>(Wup, wu_bf, nm);
        to_bf16_kernel<<<gblk(nm), 256, 0, stream>>>(Wgate, wg_bf, nm);
        to_bf16_kernel<<<gblk(nm), 256, 0, stream>>>(Wdown, wd_bf, nm);
        size_t nv = (size_t)E_ * V_;
        to_bf16_kernel<<<gblk(nv), 256, 0, stream>>>(predW, pw_bf, nv);
    }

    // ---- embedding ----
    embed_kernel<<<gblk((size_t)BS_ * E_), 256, 0, stream>>>(tokens, table, emb, embbf);

    auto gemm = [&](const unsigned short* A, const unsigned short* Bm, float* C,
                    const float* R, unsigned short* Cbf, int M, int N, int K, int mode) {
        dim3 grid(N / 128, M / 128);
        gemm_bf16_kernel<<<grid, 256, 0, stream>>>(A, Bm, C, R, Cbf, M, N, K, mode);
    };

    for (int l = 0; l < L_; ++l) {
        const unsigned short* wq_l = wq_bf + (size_t)l * E_ * 1024;
        const unsigned short* wk_l = wk_bf + (size_t)l * E_ * 1024;
        const unsigned short* wv_l = wv_bf + (size_t)l * E_ * 1024;
        const unsigned short* wp_l = wp_bf + (size_t)l * E_ * E_;
        const unsigned short* wu_l = wu_bf + (size_t)l * E_ * F_;
        const unsigned short* wg_l = wg_bf + (size_t)l * E_ * F_;
        const unsigned short* wd_l = wd_bf + (size_t)l * F_ * E_;
        const float*          nw_l = nw + (size_t)l * E_;

        // Q/K/V projections (NOTE: reference feeds RAW embeddings)
        gemm(embbf, wq_l, Qf, nullptr, nullptr, BS_, 1024, E_, 0);
        gemm(embbf, wk_l, Kf, nullptr, nullptr, BS_, 1024, E_, 0);
        gemm(embbf, wv_l, Vf, nullptr, nullptr, BS_, 1024, E_, 0);

        rope_vt_kernel<<<gblk((size_t)B_ * S_ * H_ * DK_), 256, 0, stream>>>(
            Qf, Kf, Vf, Qbf, Kbf, Vtb);

        attention_kernel<<<dim3(S_ / 16, H_, B_), 32, 0, stream>>>(Qbf, Kbf, Vtb, Obf);

        // emb = emb + O @ Wproj  (fused residual + bf16 re-emit)
        gemm(Obf, wp_l, emb, emb, embbf, BS_, E_, E_, 1);

        rmsnorm_kernel<<<BS_, 256, 0, stream>>>(emb, nw_l, hbf);

        gemm(hbf, wu_l, Uf, nullptr, nullptr, BS_, F_, E_, 0);
        gemm(hbf, wg_l, Gf, nullptr, nullptr, BS_, F_, E_, 0);

        gate_prod_kernel<<<gblk((size_t)BS_ * F_), 256, 0, stream>>>(
            Uf, Gf, pbf, (size_t)BS_ * F_);

        // emb = emb + prod @ Wdown
        gemm(pbf, wd_l, emb, emb, embbf, BS_, E_, F_, 1);
    }

    // logits = emb @ predW + pred_b
    gemm(embbf, pw_bf, out, predb, nullptr, BS_, V_, E_, 2);
}
